// GPT_48043504173506
// MI455X (gfx1250) — matrix operand order repaired and checked by reading, not compiled, not run on hardware
//
#include <hip/hip_runtime.h>
#include <math.h>

// GPT-2 forward (B=2, S=2048, E=768, H=12, D=64, FF=3072, L=4) for gfx1250.
// bf16 WMMA everywhere; weights converted+transposed to bf16 once per launch;
// activations produced directly in bf16 by upstream kernels so the GEMM inner
// loop is async-copy -> ds_load_b128 -> v_wmma with no VALU conversion.

#define VSZ 50257
#define BB  2
#define SS  2048
#define EE  768
#define HH  12
#define DD  64
#define FFN 3072
#define LLY 4
#define EPSV 1e-6f

typedef __attribute__((ext_vector_type(16))) __bf16 v16bf;
typedef __attribute__((ext_vector_type(8)))  __bf16 v8bf;
typedef __attribute__((ext_vector_type(8)))  float  v8f;
typedef __attribute__((ext_vector_type(4)))  int    v4i;

typedef __attribute__((address_space(1))) v4i g_v4i;   // global int4
typedef __attribute__((address_space(3))) v4i l_v4i;   // LDS int4

static __device__ __forceinline__ v16bf frag_from_halves(v8bf lo, v8bf hi) {
  v16bf r;
#pragma unroll
  for (int e = 0; e < 8; ++e) { r[e] = lo[e]; r[e + 8] = hi[e]; }
  return r;
}

// 16-byte global->LDS copy: async (ASYNCcnt) when available, else direct.
static __device__ __forceinline__ void async_cp16(const __bf16* gsrc, __bf16* ldst) {
#if defined(__gfx1250__) && __has_builtin(__builtin_amdgcn_global_load_async_to_lds_b128)
  __builtin_amdgcn_global_load_async_to_lds_b128(
      (g_v4i*)(void*)gsrc, (l_v4i*)ldst, 0, 0);
#else
  *(uint4*)ldst = *(const uint4*)gsrc;
#endif
}

static __device__ __forceinline__ void wait_async0() {
#if defined(__gfx1250__) && __has_builtin(__builtin_amdgcn_s_wait_asynccnt)
  __builtin_amdgcn_s_wait_asynccnt(0);
#elif defined(__gfx1250__)
  asm volatile("s_wait_asynccnt 0x0" ::: "memory");
#endif
}

// ---------------------------------------------------------------------------
// C[M,N] = A[M,K] @ B[K,N] with A bf16 row-major and BT = B^T bf16 [N,K].
// Block tile 128x128 (256 threads, 8 waves; wave tile 32x64 = 2x4 WMMA), BK=32.
// LDS rows padded to 40 bf16 (80B) so fragment ds_loads are 16B-aligned b128.
// Requires M%128==0, N%128==0, K%32==0 (true for all uses here).
// ---------------------------------------------------------------------------
__global__ __launch_bounds__(256)
void gemm_kernel(const __bf16* __restrict__ A, const __bf16* __restrict__ BT,
                 float* __restrict__ C, int M, int N, int K) {
  __shared__ __align__(16) __bf16 As[128 * 40];
  __shared__ __align__(16) __bf16 Bs[128 * 40];  // [n][k]
  const int tid  = threadIdx.x;
  const int lane = tid & 31;
  const int wave = tid >> 5;
  const int wr = (wave >> 1) * 32;   // 4 row groups
  const int wc = (wave & 1) * 64;    // 2 col groups
  const int n0 = blockIdx.x * 128;
  const int m0 = blockIdx.y * 128;

  v8f acc[2][4] = {};

  for (int kk = 0; kk < K; kk += 32) {
    // stage A (128x32) and B^T (128x32) tiles, 16B chunks per lane
#pragma unroll
    for (int c = 0; c < 2; ++c) {
      int ch = c * 256 + tid;          // 0..511
      int row = ch >> 2, col8 = (ch & 3) * 8;
      async_cp16(A + (size_t)(m0 + row) * K + kk + col8, &As[row * 40 + col8]);
      async_cp16(BT + (size_t)(n0 + row) * K + kk + col8, &Bs[row * 40 + col8]);
    }
    wait_async0();
    __syncthreads();

    v16bf a[2], b[4];
#pragma unroll
    for (int i = 0; i < 2; ++i) {
      int m = wr + i * 16 + (lane & 15);
      int kb = (lane < 16) ? 0 : 8;                     // ISA A-layout
      const __bf16* p = &As[m * 40];
      a[i] = frag_from_halves(*(const v8bf*)(p + kb), *(const v8bf*)(p + kb + 16));
    }
#pragma unroll
    for (int j = 0; j < 4; ++j) {
      int n = wc + j * 16 + (lane & 15);
      int kb = (lane < 16) ? 0 : 8;                     // ISA B-layout
      const __bf16* p = &Bs[n * 40 + kb];
      b[j] = frag_from_halves(*(const v8bf*)p, *(const v8bf*)(p + 16));
    }
#pragma unroll
    for (int i = 0; i < 2; ++i)
#pragma unroll
      for (int j = 0; j < 4; ++j)
        acc[i][j] = __builtin_amdgcn_wmma_f32_16x16x32_bf16(
            false, a[i], false, b[j], (short)0, acc[i][j], false, false);
    __syncthreads();
  }

#pragma unroll
  for (int i = 0; i < 2; ++i)
#pragma unroll
    for (int j = 0; j < 4; ++j) {
      int col = n0 + wc + j * 16 + (lane & 15);
      int rbase = m0 + wr + i * 16 + ((lane < 16) ? 0 : 8);
#pragma unroll
      for (int r = 0; r < 8; ++r)
        C[(size_t)(rbase + r) * N + col] = acc[i][j][r];
    }
}

// ---------------------------------------------------------------------------
// Flash attention: one wave per 16 queries of one (b,h); full softmax over S
// keys in chunks of 32. QK^T and P*V via bf16 WMMA; P bounced through LDS to
// convert C-layout -> A-layout. Q/K/V fp32 in, O written as bf16 (feeds the
// o-projection GEMM directly).
// ---------------------------------------------------------------------------
__global__ __launch_bounds__(32)
void attn_kernel(const float* __restrict__ Q, const float* __restrict__ Kg,
                 const float* __restrict__ Vg, __bf16* __restrict__ O) {
  __shared__ __align__(16) __bf16 Ps[16 * 40];
  const int lane = threadIdx.x;
  const int q0 = blockIdx.x * 16;
  const int h  = blockIdx.y;
  const int b  = blockIdx.z;
  const size_t base = ((size_t)b * SS) * EE + (size_t)h * DD;

  v16bf aQ[2];
  {
    int m = lane & 15;
    const float* qrow = Q + base + (size_t)(q0 + m) * EE;
#pragma unroll
    for (int t = 0; t < 2; ++t) {
      int kb = t * 32 + ((lane < 16) ? 0 : 16);
      v16bf a;
#pragma unroll
      for (int e = 0; e < 8; ++e) {
        a[e]     = (__bf16)qrow[kb + e];
        a[e + 8] = (__bf16)qrow[kb + 8 + e];
      }
      aQ[t] = a;
    }
  }

  float mrun[8], lrun[8];
  v8f oacc[4] = {};
#pragma unroll
  for (int r = 0; r < 8; ++r) { mrun[r] = -1e30f; lrun[r] = 0.f; }

  const float scale = 0.125f;  // 1/sqrt(64)

  for (int kc = 0; kc < SS; kc += 32) {
    v8f s0 = {}, s1 = {};
#pragma unroll
    for (int t = 0; t < 2; ++t) {
      int kb = t * 32 + ((lane < 16) ? 0 : 16);
      {
        const float* kr = Kg + base + (size_t)(kc + (lane & 15)) * EE;
        v16bf bk;
#pragma unroll
        for (int e = 0; e < 16; ++e) bk[e] = (__bf16)kr[kb + e];
        s0 = __builtin_amdgcn_wmma_f32_16x16x32_bf16(false, aQ[t], false, bk,
                                                     (short)0, s0, false, false);
      }
      {
        const float* kr = Kg + base + (size_t)(kc + 16 + (lane & 15)) * EE;
        v16bf bk;
#pragma unroll
        for (int e = 0; e < 16; ++e) bk[e] = (__bf16)kr[kb + e];
        s1 = __builtin_amdgcn_wmma_f32_16x16x32_bf16(false, aQ[t], false, bk,
                                                     (short)0, s1, false, false);
      }
    }

    float mnew[8], alpha[8];
#pragma unroll
    for (int r = 0; r < 8; ++r) {
      float a0 = s0[r] * scale, a1 = s1[r] * scale;
      s0[r] = a0; s1[r] = a1;
      float t = fmaxf(a0, a1);
#pragma unroll
      for (int off = 8; off; off >>= 1) t = fmaxf(t, __shfl_xor(t, off, 16));
      mnew[r]  = fmaxf(mrun[r], t);
      alpha[r] = __expf(mrun[r] - mnew[r]);
    }
#pragma unroll
    for (int r = 0; r < 8; ++r) {
      float p0 = __expf(s0[r] - mnew[r]);
      float p1 = __expf(s1[r] - mnew[r]);
      s0[r] = p0; s1[r] = p1;
      float t = p0 + p1;
#pragma unroll
      for (int off = 8; off; off >>= 1) t += __shfl_xor(t, off, 16);
      lrun[r] = lrun[r] * alpha[r] + t;
      mrun[r] = mnew[r];
    }
#pragma unroll
    for (int d = 0; d < 4; ++d)
#pragma unroll
      for (int r = 0; r < 8; ++r) oacc[d][r] *= alpha[r];

    {
      int rr = (lane < 16) ? 0 : 8;
      int c = lane & 15;
#pragma unroll
      for (int r = 0; r < 8; ++r) {
        Ps[(rr + r) * 40 + c]      = (__bf16)s0[r];
        Ps[(rr + r) * 40 + 16 + c] = (__bf16)s1[r];
      }
    }
    __syncthreads();
    v16bf aP;
    {
      int m = lane & 15;
      int kb = (lane < 16) ? 0 : 16;
      const __bf16* p = &Ps[m * 40];
      aP = frag_from_halves(*(const v8bf*)(p + kb), *(const v8bf*)(p + kb + 8));
    }

#pragma unroll
    for (int dt = 0; dt < 4; ++dt) {
      int d  = dt * 16 + (lane & 15);
      int kb = (lane < 16) ? 0 : 16;
      const float* vp = Vg + base + d;
      v16bf bv;
#pragma unroll
      for (int e = 0; e < 16; ++e)
        bv[e] = (__bf16)vp[(size_t)(kc + kb + e) * EE];
      oacc[dt] = __builtin_amdgcn_wmma_f32_16x16x32_bf16(false, aP, false, bv,
                                                         (short)0, oacc[dt], false, false);
    }
    __syncthreads();
  }

#pragma unroll
  for (int dt = 0; dt < 4; ++dt) {
    int d = dt * 16 + (lane & 15);
    int rr = (lane < 16) ? 0 : 8;
#pragma unroll
    for (int r = 0; r < 8; ++r)
      O[base + (size_t)(q0 + rr + r) * EE + d] = (__bf16)(oacc[dt][r] / lrun[r]);
  }
}

// ---------------------------------------------------------------------------
// x = LayerNorm(x [+ y]) * w ; also writes bf16 copy xb for the next GEMM.
// ---------------------------------------------------------------------------
__global__ __launch_bounds__(256)
void addln_kernel(float* __restrict__ x, const float* __restrict__ y,
                  const float* __restrict__ w, int hasY, __bf16* __restrict__ xb) {
  __shared__ float red[256];
  const int row = blockIdx.x;
  const int tid = threadIdx.x;
  float v0[3];
  float s = 0.f;
#pragma unroll
  for (int j = 0; j < 3; ++j) {
    int i = tid + j * 256;
    float t = x[(size_t)row * EE + i];
    if (hasY) t += y[(size_t)row * EE + i];
    v0[j] = t; s += t;
  }
  red[tid] = s; __syncthreads();
  for (int off = 128; off > 0; off >>= 1) {
    if (tid < off) red[tid] += red[tid + off];
    __syncthreads();
  }
  float mu = red[0] * (1.0f / EE);
  __syncthreads();
  s = 0.f;
#pragma unroll
  for (int j = 0; j < 3; ++j) { float d = v0[j] - mu; s += d * d; }
  red[tid] = s; __syncthreads();
  for (int off = 128; off > 0; off >>= 1) {
    if (tid < off) red[tid] += red[tid + off];
    __syncthreads();
  }
  float rstd = rsqrtf(red[0] * (1.0f / EE) + EPSV);
#pragma unroll
  for (int j = 0; j < 3; ++j) {
    int i = tid + j * 256;
    float o = (v0[j] - mu) * rstd * w[i];
    x[(size_t)row * EE + i] = o;
    xb[(size_t)row * EE + i] = (__bf16)o;
  }
}

__global__ __launch_bounds__(256)
void gelu_kernel(const float* __restrict__ in, __bf16* __restrict__ out, long n) {
  long i = (long)blockIdx.x * blockDim.x + threadIdx.x;
  if (i < n) {
    float v = in[i];
    out[i] = (__bf16)(0.5f * v * (1.0f + erff(v * 0.70710678118654752f)));
  }
}

__global__ __launch_bounds__(256)
void embed_kernel(const int* __restrict__ idx, const float* __restrict__ wte,
                  const float* __restrict__ wpe, float* __restrict__ x,
                  __bf16* __restrict__ xb) {
  int row = blockIdx.x;
  int tid = threadIdx.x;
  int tok = idx[row];
  int s = row & (SS - 1);
#pragma unroll
  for (int j = 0; j < 3; ++j) {
    int i = tid + j * 256;
    float t = wte[(size_t)tok * EE + i] + wpe[(size_t)s * EE + i];
    x[(size_t)row * EE + i] = t;
    xb[(size_t)row * EE + i] = (__bf16)t;
  }
}

// out[N,K] (bf16) = transpose of in[K,N] (fp32); 32x32 LDS tiles.
__global__ __launch_bounds__(256)
void transpose_f2b(const float* __restrict__ in, __bf16* __restrict__ out,
                   int K, int N) {
  __shared__ float t[32][33];
  int n0 = blockIdx.x * 32, k0 = blockIdx.y * 32;
  int tx = threadIdx.x, ty = threadIdx.y;
#pragma unroll
  for (int j = 0; j < 32; j += 8)
    t[ty + j][tx] = in[(size_t)(k0 + ty + j) * N + n0 + tx];
  __syncthreads();
#pragma unroll
  for (int j = 0; j < 32; j += 8)
    out[(size_t)(n0 + ty + j) * K + k0 + tx] = (__bf16)t[tx][ty + j];
}

// logits[b, v] = dot(x[b, S-1, :], wte[v, :]) — one wave per vocab row
__global__ __launch_bounds__(256)
void logits_kernel(const float* __restrict__ x, const float* __restrict__ wte,
                   float* __restrict__ out) {
  int wave = threadIdx.x >> 5, lane = threadIdx.x & 31;
  int v = blockIdx.x * 8 + wave;
  int b = blockIdx.y;
  if (v >= VSZ) return;
  const float* xr = x + ((size_t)b * SS + (SS - 1)) * EE;
  const float* wv = wte + (size_t)v * EE;
  float s = 0.f;
  for (int i = lane; i < EE; i += 32) s += xr[i] * wv[i];
#pragma unroll
  for (int off = 16; off; off >>= 1) s += __shfl_xor(s, off, 32);
  if (lane == 0) out[(size_t)b * VSZ + v] = s;
}

// ---------------------------------------------------------------------------
extern "C" void kernel_launch(void* const* d_in, const int* in_sizes, int n_in,
                              void* d_out, int out_size, void* d_ws, size_t ws_size,
                              hipStream_t stream) {
  const int*   idx  = (const int*)d_in[0];
  const float* wte  = (const float*)d_in[1];
  const float* wpe  = (const float*)d_in[2];
  const float* wq   = (const float*)d_in[3];
  const float* wk   = (const float*)d_in[4];
  const float* wv   = (const float*)d_in[5];
  const float* wo   = (const float*)d_in[6];
  const float* ln1  = (const float*)d_in[7];
  const float* ln2  = (const float*)d_in[8];
  const float* win  = (const float*)d_in[9];
  const float* wout = (const float*)d_in[10];
  const float* lnf  = (const float*)d_in[11];
  float* out = (float*)d_out;

  const size_t sXE = (size_t)BB * SS * EE;    // 3,145,728
  const size_t sFF = (size_t)BB * SS * FFN;   // 12,582,912
  const size_t sEE = (size_t)EE * EE;
  const size_t sEF = (size_t)EE * FFN;
  const size_t LW  = 4 * sEE + 2 * sEF;       // bf16 weights per layer

  float* ws = (float*)d_ws;
  float* x  = ws;               // fp32 activations
  float* q  = ws + 1 * sXE;
  float* k  = ws + 2 * sXE;
  float* v  = ws + 3 * sXE;
  float* h1 = ws + 4 * sXE;     // [M,FF] MLP hidden fp32
  float* y  = q;                // alias: free after attention

  __bf16* bbase = (__bf16*)(ws + 4 * sXE + sFF);
  __bf16* xb = bbase;               // bf16 activations
  __bf16* tb = bbase + sXE;         // bf16 attention output
  __bf16* hb = bbase + 2 * sXE;     // bf16 GELU output
  __bf16* wT = bbase + 2 * sXE + sFF;  // transposed bf16 weights, all layers

  const int M = BB * SS;  // 4096
  const dim3 tb32(32, 8);

  // weight convert+transpose (fp32 [K,N] -> bf16 [N,K]) once per launch
  for (int l = 0; l < LLY; ++l) {
    __bf16* wl = wT + (size_t)l * LW;
    transpose_f2b<<<dim3(EE / 32, EE / 32), tb32, 0, stream>>>(wq + (size_t)l * sEE, wl + 0 * sEE, EE, EE);
    transpose_f2b<<<dim3(EE / 32, EE / 32), tb32, 0, stream>>>(wk + (size_t)l * sEE, wl + 1 * sEE, EE, EE);
    transpose_f2b<<<dim3(EE / 32, EE / 32), tb32, 0, stream>>>(wv + (size_t)l * sEE, wl + 2 * sEE, EE, EE);
    transpose_f2b<<<dim3(EE / 32, EE / 32), tb32, 0, stream>>>(wo + (size_t)l * sEE, wl + 3 * sEE, EE, EE);
    transpose_f2b<<<dim3(FFN / 32, EE / 32), tb32, 0, stream>>>(win + (size_t)l * sEF, wl + 4 * sEE, EE, FFN);
    transpose_f2b<<<dim3(EE / 32, FFN / 32), tb32, 0, stream>>>(wout + (size_t)l * sEF, wl + 4 * sEE + sEF, FFN, EE);
  }

  embed_kernel<<<M, 256, 0, stream>>>(idx, wte, wpe, x, xb);

  for (int l = 0; l < LLY; ++l) {
    __bf16* wl = wT + (size_t)l * LW;
    __bf16* qT = wl + 0 * sEE;
    __bf16* kT = wl + 1 * sEE;
    __bf16* vT = wl + 2 * sEE;
    __bf16* oT = wl + 3 * sEE;
    __bf16* iT = wl + 4 * sEE;
    __bf16* uT = wl + 4 * sEE + sEF;

    gemm_kernel<<<dim3(EE / 128, M / 128), 256, 0, stream>>>(xb, qT, q, M, EE, EE);
    gemm_kernel<<<dim3(EE / 128, M / 128), 256, 0, stream>>>(xb, kT, k, M, EE, EE);
    gemm_kernel<<<dim3(EE / 128, M / 128), 256, 0, stream>>>(xb, vT, v, M, EE, EE);

    attn_kernel<<<dim3(SS / 16, HH, BB), 32, 0, stream>>>(q, k, v, tb);

    gemm_kernel<<<dim3(EE / 128, M / 128), 256, 0, stream>>>(tb, oT, y, M, EE, EE);
    addln_kernel<<<M, 256, 0, stream>>>(x, y, ln1 + (size_t)l * EE, 1, xb);

    gemm_kernel<<<dim3(FFN / 128, M / 128), 256, 0, stream>>>(xb, iT, h1, M, FFN, EE);
    gelu_kernel<<<(int)(((long)M * FFN + 255) / 256), 256, 0, stream>>>(h1, hb, (long)M * FFN);
    gemm_kernel<<<dim3(EE / 128, M / 128), 256, 0, stream>>>(hb, uT, y, M, EE, FFN);
    addln_kernel<<<M, 256, 0, stream>>>(x, y, ln2 + (size_t)l * EE, 1, xb);
  }

  addln_kernel<<<M, 256, 0, stream>>>(x, nullptr, lnf, 0, xb);
  logits_kernel<<<dim3((VSZ + 7) / 8, BB), 256, 0, stream>>>(x, wte, out);
}
